// CrossAttentionPositionBridge_70772471104044
// MI455X (gfx1250) — matrix operand order repaired and checked by reading, not compiled, not run on hardware
//
#include <hip/hip_runtime.h>
#include <hip/hip_bf16.h>

// ---------------------------------------------------------------------------
// CrossAttentionPositionBridge for MI455X (gfx1250, wave32).
// GEMM chain (6x [16384x1024]x[1024x1024]) runs on v_wmma_f32_16x16x32_bf16
// with f32 accumulation. The shared B (weight) tile is staged through LDS
// via global_load_async_to_lds_b128 (ASYNCcnt path) so each block loads it
// once instead of 8x. Segmented softmax glue stays in f32 with atomics.
// ---------------------------------------------------------------------------

typedef __attribute__((ext_vector_type(16))) __bf16 v16bf;
typedef __attribute__((ext_vector_type(8)))  __bf16 v8bf;
typedef __attribute__((ext_vector_type(8)))  float  v8f;
typedef int v4i __attribute__((vector_size(16)));

#define AS1 __attribute__((address_space(1)))
#define AS3 __attribute__((address_space(3)))

#define NB   4
#define SS   4096
#define DIM  1024
#define NH   16
#define HD   64
#define NTOK (NB * SS)            // 16384 flat positions / segment-id upper bound

#if __has_builtin(__builtin_amdgcn_global_load_async_to_lds_b128)
#define HAVE_ASYNC_LDS 1
#endif

__device__ __forceinline__ unsigned short f32_to_bf16(float f) {
  unsigned u = __float_as_uint(f);
  u += 0x7FFFu + ((u >> 16) & 1u);          // round-to-nearest-even
  return (unsigned short)(u >> 16);
}
__device__ __forceinline__ float bf16_to_f32(unsigned short h) {
  return __uint_as_float(((unsigned)h) << 16);
}
// monotone float -> uint mapping so unsigned atomicMax == float max
__device__ __forceinline__ unsigned fmap(float f) {
  unsigned u = __float_as_uint(f);
  return (u & 0x80000000u) ? ~u : (u | 0x80000000u);
}
__device__ __forceinline__ float funmap(unsigned u) {
  unsigned b = (u & 0x80000000u) ? (u & 0x7FFFFFFFu) : ~u;
  return __uint_as_float(b);
}

// 16-byte global -> LDS stage: async copy (ASYNCcnt) when available.
__device__ __forceinline__ void stage16(const unsigned short* gp, unsigned short* lp) {
#ifdef HAVE_ASYNC_LDS
  __builtin_amdgcn_global_load_async_to_lds_b128(
      (AS1 v4i*)gp, (AS3 v4i*)lp, 0, 0);
#else
  *reinterpret_cast<v8bf*>(lp) = *reinterpret_cast<const v8bf*>(gp);
#endif
}
__device__ __forceinline__ void stage_wait() {
#ifdef HAVE_ASYNC_LDS
#if __has_builtin(__builtin_amdgcn_s_wait_asynccnt)
  __builtin_amdgcn_s_wait_asynccnt(0);
#else
  asm volatile("s_wait_asynccnt 0" ::: "memory");
#endif
#endif
}

// ---------------- elementwise helpers ----------------
__global__ void zero_u32_kernel(unsigned* __restrict__ p, size_t n) {
  size_t i = (size_t)blockIdx.x * 256 + threadIdx.x;
  if (i < n) p[i] = 0u;
}
__global__ void cvt_bf16_kernel(const float* __restrict__ s,
                                unsigned short* __restrict__ d, size_t n) {
  size_t i = (size_t)blockIdx.x * 256 + threadIdx.x;
  if (i < n) d[i] = f32_to_bf16(s[i]);
}

// ---------------- segment-id scan (one block per batch row) ----------------
__global__ void segscan_kernel(const int* __restrict__ bnd,
                               int* __restrict__ seg,
                               float* __restrict__ cnt) {
  __shared__ int partial[256];
  const int b = blockIdx.x;
  const int t = threadIdx.x;
  const int CH = SS / 256;                       // 16 elements per thread
  const int* row = bnd + (size_t)b * SS;
  const int base = t * CH;
  int flags[CH];
  int s = 0;
  #pragma unroll
  for (int e = 0; e < CH; ++e) { flags[e] = (row[base + e] != 0) ? 1 : 0; s += flags[e]; }
  partial[t] = s;
  __syncthreads();
  for (int off = 1; off < 256; off <<= 1) {      // Hillis-Steele inclusive scan
    int v = partial[t];
    int a = (t >= off) ? partial[t - off] : 0;
    __syncthreads();
    partial[t] = v + a;
    __syncthreads();
  }
  int run = partial[t] - s;                      // exclusive prefix
  const int b0 = (row[0] != 0) ? 1 : 0;          // reference: cumsum - bnd[:, :1]
  for (int e = 0; e < CH; ++e) {
    run += flags[e];
    int sg = (run - b0) + b * SS;
    seg[(size_t)b * SS + base + e] = sg;
    atomicAdd(&cnt[sg], 1.0f);
  }
}

// ---------------- segment sum of x (f32 atomics into scratch F) ------------
__global__ void xsum_kernel(const float* __restrict__ x,
                            const int* __restrict__ seg,
                            float* __restrict__ F) {
  size_t i = (size_t)blockIdx.x * 256 + threadIdx.x;   // over NTOK*DIM
  int n = (int)(i >> 10);
  int d = (int)(i & 1023);
  atomicAdd(&F[(size_t)seg[n] * DIM + d], x[i]);
}

__global__ void qmean_kernel(const float* __restrict__ F,
                             const float* __restrict__ cnt,
                             unsigned short* __restrict__ Qm) {
  size_t i = (size_t)blockIdx.x * 256 + threadIdx.x;   // over NTOK*DIM
  int s = (int)(i >> 10);
  float c = cnt[s];
  c = c < 1.0f ? 1.0f : c;
  Qm[i] = f32_to_bf16(F[i] / c);
}

// ---------------- WMMA bf16 GEMM:  out[m,c] = sum_k A[rm(m),k]*W[c,k] + bias[c]
__device__ __forceinline__ v16bf ld_a_frag(const unsigned short* p) {
  // ISA 16-bit A 16x32 layout: lanes 0-15 hold K[kb..kb+7] (v0-3) and
  // K[kb+16..kb+23] (v4-7); lanes 16-31 the +8 halves.
  union { v16bf v; v8bf h[2]; } u;
  u.h[0] = *reinterpret_cast<const v8bf*>(p);
  u.h[1] = *reinterpret_cast<const v8bf*>(p + 16);
  return u.v;
}

template <bool OUT_BF16>
__global__ __launch_bounds__(256) void gemm_bf16_kernel(
    const unsigned short* __restrict__ A,   // [M,K] bf16 row-major (via rowmap)
    const unsigned short* __restrict__ W,   // [Nc,K] bf16 row-major (y = x @ W^T)
    const float* __restrict__ bias,         // [Nc]
    void* __restrict__ outp,                // [M,Nc] f32 or bf16
    const int* __restrict__ rowmap,         // optional A-row gather
    int M, int K, int Nc) {
  constexpr int KC   = 128;                 // K-chunk staged in LDS
  constexpr int LSTR = 136;                 // ushorts per W row (272 B: +16 B pad
                                            // -> 16 fragment lanes hit 16 banks)
  __shared__ unsigned short Bs[64 * LSTR];  // 17.4 KB

  const int lane = threadIdx.x & 31;
  const int w    = threadIdx.x >> 5;        // 8 waves, 32 rows each
  const int mblocks = M >> 8;               // 256-row block tiles
  const int bm = blockIdx.x % mblocks;
  const int bn = blockIdx.x / mblocks;      // block owns one 64-column tile
  const int row0 = (bm << 8) + (w << 5);
  const int col0 = bn << 6;
  const int lr    = lane & 15;
  const int akoff = (lane >> 4) << 3;       // 0 / 8
  const int bkoff = (lane >> 4) << 4;       // 0 / 16

  int ar0 = row0 + lr, ar1 = row0 + 16 + lr;
  if (rowmap) { ar0 = rowmap[ar0]; ar1 = rowmap[ar1]; }
  const unsigned short* ap0 = A + (size_t)ar0 * K + akoff;
  const unsigned short* ap1 = A + (size_t)ar1 * K + akoff;

  // cooperative staging coords: thread -> (W row, 64-byte span of the chunk)
  const int srow = threadIdx.x >> 2;                    // 0..63
  const int sx   = (threadIdx.x & 3) << 5;              // 0,32,64,96 elements
  const unsigned short* sgp = W + (size_t)(col0 + srow) * K + sx;
  unsigned short*       slp = &Bs[srow * LSTR + sx];

  v8f acc[2][4];
  #pragma unroll
  for (int i = 0; i < 2; ++i)
    #pragma unroll
    for (int j = 0; j < 4; ++j)
      acc[i][j] = (v8f){0.f, 0.f, 0.f, 0.f, 0.f, 0.f, 0.f, 0.f};

  for (int kc = 0; kc < K; kc += KC) {
    #pragma unroll
    for (int i = 0; i < 4; ++i)                    // 64 B per thread per chunk
      stage16(sgp + kc + i * 8, slp + i * 8);
    stage_wait();
    __syncthreads();
    #pragma unroll
    for (int kb = 0; kb < KC; kb += 32) {
      v16bf a0 = ld_a_frag(ap0 + kc + kb);
      v16bf a1 = ld_a_frag(ap1 + kc + kb);
      #pragma unroll
      for (int j = 0; j < 4; ++j) {
        const unsigned short* bl = &Bs[((j << 4) + lr) * LSTR + kb + akoff];
        union { v16bf v; v8bf h[2]; } ub;          // 2x ds_load_b128
        ub.h[0] = *reinterpret_cast<const v8bf*>(bl);
        ub.h[1] = *reinterpret_cast<const v8bf*>(bl + 16);
        acc[0][j] = __builtin_amdgcn_wmma_f32_16x16x32_bf16(
            false, a0, false, ub.v, (short)0, acc[0][j], false, false);
        acc[1][j] = __builtin_amdgcn_wmma_f32_16x16x32_bf16(
            false, a1, false, ub.v, (short)0, acc[1][j], false, false);
      }
    }
    __syncthreads();
  }

  float*          of = (float*)outp;
  unsigned short* ob = (unsigned short*)outp;
  #pragma unroll
  for (int i = 0; i < 2; ++i) {
    #pragma unroll
    for (int j = 0; j < 4; ++j) {
      const int c  = col0 + (j << 4) + lr;
      const float bv = bias ? bias[c] : 0.f;
      #pragma unroll
      for (int e = 0; e < 8; ++e) {
        const int r = row0 + (i << 4) + e + akoff;   // C layout: M = e + 8*(lane>>4)
        const size_t o = (size_t)r * Nc + c;
        const float val = acc[i][j][e] + bv;
        if (OUT_BF16) ob[o] = f32_to_bf16(val);
        else          of[o] = val;
      }
    }
  }
}

// ---------------- attention glue (f32) ----------------
__global__ void score_kernel(const unsigned short* __restrict__ Qb,
                             const unsigned short* __restrict__ Kb,
                             const int* __restrict__ seg,
                             float* __restrict__ score,
                             unsigned* __restrict__ mU) {
  int idx = blockIdx.x * 256 + threadIdx.x;        // over NTOK*NH
  int n = idx >> 4, h = idx & 15;
  int s = seg[n];
  const unsigned* qp = (const unsigned*)(Qb + (size_t)s * DIM + h * HD);
  const unsigned* kp = (const unsigned*)(Kb + (size_t)n * DIM + h * HD);
  float acc = 0.f;
  #pragma unroll
  for (int t = 0; t < HD / 2; ++t) {
    unsigned qv = qp[t], kv = kp[t];
    acc += bf16_to_f32((unsigned short)(qv & 0xFFFFu)) * bf16_to_f32((unsigned short)(kv & 0xFFFFu));
    acc += bf16_to_f32((unsigned short)(qv >> 16))     * bf16_to_f32((unsigned short)(kv >> 16));
  }
  acc *= 0.125f;                                    // 1/sqrt(64)
  score[idx] = acc;
  atomicMax(&mU[s * NH + h], fmap(acc));
}

__global__ void psum_kernel(float* __restrict__ score,
                            const unsigned* __restrict__ mU,
                            const int* __restrict__ seg,
                            float* __restrict__ denom) {
  int idx = blockIdx.x * 256 + threadIdx.x;        // over NTOK*NH
  int n = idx >> 4, h = idx & 15;
  int s = seg[n];
  float p = expf(score[idx] - funmap(mU[s * NH + h]));
  score[idx] = p;
  atomicAdd(&denom[s * NH + h], p);
}

__global__ void phacc_kernel(const float* __restrict__ p,
                             const float* __restrict__ denom,
                             const unsigned short* __restrict__ Vb,
                             const int* __restrict__ seg,
                             float* __restrict__ F) {
  size_t i = (size_t)blockIdx.x * 256 + threadIdx.x;   // over NTOK*DIM
  int n = (int)(i >> 10);
  int d = (int)(i & 1023);
  int h = d >> 6;
  int s = seg[n];
  float w = p[(size_t)n * NH + h] / denom[s * NH + h];
  atomicAdd(&F[(size_t)s * DIM + d], w * bf16_to_f32(Vb[i]));
}

// ---------------------------------------------------------------------------
extern "C" void kernel_launch(void* const* d_in, const int* in_sizes, int n_in,
                              void* d_out, int out_size, void* d_ws, size_t ws_size,
                              hipStream_t stream) {
  (void)in_sizes; (void)n_in; (void)out_size; (void)ws_size;
  const float* x   = (const float*)d_in[0];
  const float* Wq  = (const float*)d_in[1];  const float* bq  = (const float*)d_in[2];
  const float* Wk  = (const float*)d_in[3];  const float* bk  = (const float*)d_in[4];
  const float* Wv  = (const float*)d_in[5];  const float* bv  = (const float*)d_in[6];
  const float* Wo  = (const float*)d_in[7];  const float* bo  = (const float*)d_in[8];
  const float* Wv2 = (const float*)d_in[9];  const float* bv2 = (const float*)d_in[10];
  const float* Wo2 = (const float*)d_in[11]; const float* bo2 = (const float*)d_in[12];
  const int*   bnd = (const int*)d_in[13];

  const size_t ND = (size_t)NTOK * DIM;            // 16.7M elements
  const size_t DD = (size_t)DIM * DIM;

  char* ws = (char*)d_ws;
  size_t off = 0;
  auto alloc = [&](size_t bytes) -> void* {
    void* p = ws + off;
    off += (bytes + 255) & ~(size_t)255;
    return p;
  };
  int*            seg   = (int*)           alloc((size_t)NTOK * 4);
  float*          cnt   = (float*)         alloc((size_t)NTOK * 4);
  unsigned*       mU    = (unsigned*)      alloc((size_t)NTOK * NH * 4);
  float*          denom = (float*)         alloc((size_t)NTOK * NH * 4);
  float*          score = (float*)         alloc((size_t)NTOK * NH * 4);
  float*          F     = (float*)         alloc(ND * 4);   // xsum, then patch_heads
  unsigned short* Xb    = (unsigned short*)alloc(ND * 2);   // x bf16
  unsigned short* Qm    = (unsigned short*)alloc(ND * 2);   // qmean bf16
  unsigned short* Qb    = (unsigned short*)alloc(ND * 2);   // q bf16 -> patch_heads bf16
  unsigned short* Kb    = (unsigned short*)alloc(ND * 2);   // k bf16 -> patch bf16
  unsigned short* Vb    = (unsigned short*)alloc(ND * 2);   // v bf16 -> pv bf16
  unsigned short* Wb    = (unsigned short*)alloc(6 * DD * 2);

  auto g = [](size_t n) { return (unsigned)((n + 255) / 256); };

  // --- init accumulators ---
  zero_u32_kernel<<<g(NTOK), 256, 0, stream>>>((unsigned*)cnt, NTOK);
  zero_u32_kernel<<<g((size_t)NTOK * NH), 256, 0, stream>>>(mU, (size_t)NTOK * NH);
  zero_u32_kernel<<<g((size_t)NTOK * NH), 256, 0, stream>>>((unsigned*)denom, (size_t)NTOK * NH);
  zero_u32_kernel<<<g(ND), 256, 0, stream>>>((unsigned*)F, ND);

  // --- segment ids + counts ---
  segscan_kernel<<<NB, 256, 0, stream>>>(bnd, seg, cnt);

  // --- bf16 conversions (x + 6 weights) ---
  cvt_bf16_kernel<<<g(ND), 256, 0, stream>>>(x, Xb, ND);
  cvt_bf16_kernel<<<g(DD), 256, 0, stream>>>(Wq,  Wb + 0 * DD, DD);
  cvt_bf16_kernel<<<g(DD), 256, 0, stream>>>(Wk,  Wb + 1 * DD, DD);
  cvt_bf16_kernel<<<g(DD), 256, 0, stream>>>(Wv,  Wb + 2 * DD, DD);
  cvt_bf16_kernel<<<g(DD), 256, 0, stream>>>(Wo,  Wb + 3 * DD, DD);
  cvt_bf16_kernel<<<g(DD), 256, 0, stream>>>(Wv2, Wb + 4 * DD, DD);
  cvt_bf16_kernel<<<g(DD), 256, 0, stream>>>(Wo2, Wb + 5 * DD, DD);

  // --- qmean = segment_sum(x)/cnt ---
  xsum_kernel<<<g(ND), 256, 0, stream>>>(x, seg, F);
  qmean_kernel<<<g(ND), 256, 0, stream>>>(F, cnt, Qm);
  zero_u32_kernel<<<g(ND), 256, 0, stream>>>((unsigned*)F, ND);  // reuse F for patch_heads

  // --- q/k/v projections on WMMA ---
  const unsigned gemmBlocks = (NTOK / 256) * (DIM / 64);         // 1024 blocks, 8 waves each
  gemm_bf16_kernel<true><<<gemmBlocks, 256, 0, stream>>>(Qm, Wb + 0 * DD, bq, Qb, nullptr, NTOK, DIM, DIM);
  gemm_bf16_kernel<true><<<gemmBlocks, 256, 0, stream>>>(Xb, Wb + 1 * DD, bk, Kb, nullptr, NTOK, DIM, DIM);
  gemm_bf16_kernel<true><<<gemmBlocks, 256, 0, stream>>>(Xb, Wb + 2 * DD, bv, Vb, nullptr, NTOK, DIM, DIM);

  // --- segmented softmax over per-byte scores ---
  score_kernel<<<g((size_t)NTOK * NH), 256, 0, stream>>>(Qb, Kb, seg, score, mU);
  psum_kernel <<<g((size_t)NTOK * NH), 256, 0, stream>>>(score, mU, seg, denom);
  phacc_kernel<<<g(ND), 256, 0, stream>>>(score, denom, Vb, seg, F);

  // --- patch -> pv -> byte_out on WMMA (final GEMM gathers pv[seg]) ---
  cvt_bf16_kernel<<<g(ND), 256, 0, stream>>>(F, Qb, ND);          // patch_heads bf16 (reuse Qb)
  gemm_bf16_kernel<true ><<<gemmBlocks, 256, 0, stream>>>(Qb, Wb + 3 * DD, bo,  Kb, nullptr, NTOK, DIM, DIM);
  gemm_bf16_kernel<true ><<<gemmBlocks, 256, 0, stream>>>(Kb, Wb + 4 * DD, bv2, Vb, nullptr, NTOK, DIM, DIM);
  gemm_bf16_kernel<false><<<gemmBlocks, 256, 0, stream>>>(Vb, Wb + 5 * DD, bo2, d_out, seg, NTOK, DIM, DIM);
}
